// ColumnBlock_67199058313496
// MI455X (gfx1250) — matrix operand order repaired and checked by reading, not compiled, not run on hardware
//
#include <hip/hip_runtime.h>
#include <hip/hip_bf16.h>

typedef __bf16 bf16_t;
typedef __attribute__((ext_vector_type(16))) __bf16 v16bf;
typedef __attribute__((ext_vector_type(8)))  float  v8f;

constexpr int kB = 8, kN = 2048, kC = 8, kD = 128;
constexpr int kNC = kN * kC;          // 16384 flattened (n,c) rows per batch
constexpr int kR = 64, kM = 4096, kCEM = 32;
constexpr float kTauInv = 10.0f;      // 1 / TAU_ROUTE_PM
constexpr float kEps = 1e-8f;

// ---- WMMA fragment helpers (bf16 16x16x32, wave32) -------------------------
// A (16x32, row-major source [row][k], ld elements): lane L<16 -> row=row0+L,
// K = k0..k0+7 (v0..3) and k0+16..k0+23 (v4..7); lanes 16..31 hold K+8.
__device__ __forceinline__ v16bf load_frag_a(const bf16_t* base, int ld,
                                             int row0, int k0, int lane) {
  int row = row0 + (lane & 15);
  int kk  = k0 + ((lane >> 4) << 3);
  const bf16_t* p = base + row * ld + kk;
  v16bf a;
  reinterpret_cast<uint4*>(&a)[0] = *reinterpret_cast<const uint4*>(p);
  reinterpret_cast<uint4*>(&a)[1] = *reinterpret_cast<const uint4*>(p + 16);
  return a;
}
// B (32x16, source stored column(N)-major [n][k], K contiguous): lane L<16 ->
// col=col0+L, K = k0..k0+7 (v0..3) and k0+16..k0+23 (v4..7); lanes 16..31 hold K+8.
__device__ __forceinline__ v16bf load_frag_b(const bf16_t* base, int ld,
                                             int col0, int k0, int lane) {
  int col = col0 + (lane & 15);
  int kk  = k0 + ((lane >> 4) << 3);
  const bf16_t* p = base + col * ld + kk;
  v16bf b;
  reinterpret_cast<uint4*>(&b)[0] = *reinterpret_cast<const uint4*>(p);
  reinterpret_cast<uint4*>(&b)[1] = *reinterpret_cast<const uint4*>(p + 16);
  return b;
}
__device__ __forceinline__ v8f wmma_bf16(v16bf a, v16bf b, v8f c) {
  return __builtin_amdgcn_wmma_f32_16x16x32_bf16(false, a, false, b,
                                                 (short)0, c, false, false);
}

// ===========================================================================
// Kernel 1: fused PM routing + eligibility accumulation.
// One workgroup = one batch b and a 128-row tile of the flattened (n,c) dim.
// ===========================================================================
__global__ __launch_bounds__(256) void k_route_elig(
    const float* __restrict__ k_cand, const float* __restrict__ v_cand,
    const float* __restrict__ gate,   const float* __restrict__ pm_K,
    float* __restrict__ eligK, float* __restrict__ eligV) {
  __shared__ alignas(16) bf16_t kn[128 * 128];      // normalized k, row-major
  __shared__ alignas(16) bf16_t krawT[128 * 128];   // raw k, [d][row]
  __shared__ alignas(16) bf16_t vrawT[128 * 128];   // raw v, [d][row]
  __shared__ alignas(16) bf16_t pmb[64 * 128];      // pm_K bf16, [r][d]
  __shared__ alignas(16) bf16_t routedT[64 * 128];  // gated route_w, [r][row]
  __shared__ float scores[128 * 65];                // padded to dodge bank conflicts
  __shared__ float gates[128];

  const int b   = blockIdx.x >> 7;
  const int nc0 = (blockIdx.x & 127) * 128;
  const int tid = threadIdx.x;
  const int lane = tid & 31;
  const int wave = tid >> 5;

  // ---- Phase 1: stage tiles (2 threads per row; partner reduce for norm) ---
  {
    const int row = tid >> 1, half = tid & 1;
    const float* src = k_cand + ((size_t)b * kNC + nc0 + row) * kD + half * 64;
    float v[64];
    float ss = 0.f;
#pragma unroll
    for (int i = 0; i < 16; ++i) {
      float4 f = reinterpret_cast<const float4*>(src)[i];
      v[i * 4 + 0] = f.x; v[i * 4 + 1] = f.y;
      v[i * 4 + 2] = f.z; v[i * 4 + 3] = f.w;
      ss += f.x * f.x + f.y * f.y + f.z * f.z + f.w * f.w;
    }
    ss += __shfl_xor(ss, 1, 32);
    const float scale = 1.0f / (sqrtf(ss) + kEps);
#pragma unroll
    for (int i = 0; i < 64; ++i) {
      const int d = half * 64 + i;
      kn[row * 128 + d]   = (bf16_t)(v[i] * scale);
      krawT[d * 128 + row] = (bf16_t)v[i];
    }
    const float* vs = v_cand + ((size_t)b * kNC + nc0 + row) * kD + half * 64;
#pragma unroll
    for (int i = 0; i < 64; ++i)
      vrawT[(half * 64 + i) * 128 + row] = (bf16_t)vs[i];
  }
  for (int i = tid; i < kR * kD; i += 256)
    pmb[i] = (bf16_t)pm_K[(size_t)b * kR * kD + i];
  if (tid < 128) gates[tid] = gate[(size_t)b * kNC + nc0 + tid];
  __syncthreads();

  // ---- Phase 2: route_scores = kn(128x128) @ pm^T(128x64) via WMMA --------
  {
    const v8f z = {0.f, 0.f, 0.f, 0.f, 0.f, 0.f, 0.f, 0.f};
    v8f acc[4] = {z, z, z, z};
#pragma unroll
    for (int kk = 0; kk < 4; ++kk) {
      v16bf a = load_frag_a(kn, 128, wave * 16, kk * 32, lane);
#pragma unroll
      for (int ct = 0; ct < 4; ++ct)
        acc[ct] = wmma_bf16(a, load_frag_b(pmb, 128, ct * 16, kk * 32, lane),
                            acc[ct]);
    }
#pragma unroll
    for (int ct = 0; ct < 4; ++ct)
#pragma unroll
      for (int i = 0; i < 8; ++i) {
        const int row = wave * 16 + ((lane >> 4) << 3) + i;
        const int col = ct * 16 + (lane & 15);
        scores[row * 65 + col] = acc[ct][i];
      }
  }
  __syncthreads();

  // ---- Phase 3: softmax over r (tau=0.1), gate, write routed^T bf16 -------
  if (tid < 128) {
    float m = -3.0e38f;
    for (int j = 0; j < kR; ++j) m = fmaxf(m, scores[tid * 65 + j]);
    float s = 0.f;
    for (int j = 0; j < kR; ++j) {
      float e = __expf((scores[tid * 65 + j] - m) * kTauInv);
      scores[tid * 65 + j] = e;
      s += e;
    }
    const float g = gates[tid] / s;
    for (int j = 0; j < kR; ++j)
      routedT[j * 128 + tid] = (bf16_t)(scores[tid * 65 + j] * g);
  }
  __syncthreads();

  // ---- Phase 4: elig += routed^T(64x128) @ {k,v}(128x128), f32 atomics ----
  {
    const int rt = (wave & 3) * 16;   // r-tile
    const int cb = (wave >> 2) * 4;   // d-tile base (4 tiles per wave)
    const v8f z = {0.f, 0.f, 0.f, 0.f, 0.f, 0.f, 0.f, 0.f};
    v8f accK[4] = {z, z, z, z}, accV[4] = {z, z, z, z};
#pragma unroll
    for (int kk = 0; kk < 4; ++kk) {
      v16bf a = load_frag_a(routedT, 128, rt, kk * 32, lane);
#pragma unroll
      for (int j = 0; j < 4; ++j) {
        accK[j] = wmma_bf16(a, load_frag_b(krawT, 128, (cb + j) * 16, kk * 32, lane), accK[j]);
        accV[j] = wmma_bf16(a, load_frag_b(vrawT, 128, (cb + j) * 16, kk * 32, lane), accV[j]);
      }
    }
#pragma unroll
    for (int j = 0; j < 4; ++j)
#pragma unroll
      for (int i = 0; i < 8; ++i) {
        const int r = rt + ((lane >> 4) << 3) + i;
        const int d = (cb + j) * 16 + (lane & 15);
        unsafeAtomicAdd(&eligK[((size_t)b * kR + r) * kD + d], accK[j][i]);
        unsafeAtomicAdd(&eligV[((size_t)b * kR + r) * kD + d], accV[j][i]);
      }
  }
}

// ===========================================================================
// Kernel 2: normalize em_K rows -> bf16 workspace (one wave per row).
// ===========================================================================
__global__ __launch_bounds__(256) void k_em_norm(const float* __restrict__ em_K,
                                                 bf16_t* __restrict__ en) {
  const int lane = threadIdx.x & 31, wave = threadIdx.x >> 5;
  const size_t row = (size_t)blockIdx.x * 8 + wave;   // < kB*kM
  float4 f = *reinterpret_cast<const float4*>(em_K + row * kD + lane * 4);
  float ss = f.x * f.x + f.y * f.y + f.z * f.z + f.w * f.w;
#pragma unroll
  for (int o = 1; o < 32; o <<= 1) ss += __shfl_xor(ss, o, 32);
  const float scale = 1.0f / (sqrtf(ss) + kEps);
  struct alignas(8) B4 { bf16_t v[4]; } o;
  o.v[0] = (bf16_t)(f.x * scale); o.v[1] = (bf16_t)(f.y * scale);
  o.v[2] = (bf16_t)(f.z * scale); o.v[3] = (bf16_t)(f.w * scale);
  *reinterpret_cast<B4*>(en + row * kD + lane * 4) = o;
}

// ===========================================================================
// Kernel 3: novelty = w*surp*(1 - max_m <qn, en_m>). sim via WMMA, max kept
// in registers; sim matrix never touches memory.
// ===========================================================================
__global__ __launch_bounds__(256) void k_novelty(
    const float* __restrict__ q_nov, const bf16_t* __restrict__ en,
    const float* __restrict__ w_nov, const float* __restrict__ surp,
    float* __restrict__ novelty) {
  __shared__ alignas(16) bf16_t qn[128 * 128];
  const int b  = blockIdx.x >> 4;
  const int n0 = (blockIdx.x & 15) * 128;
  const int tid = threadIdx.x, lane = tid & 31, wave = tid >> 5;

  {  // normalize q tile into LDS
    const int row = tid >> 1, half = tid & 1;
    const float* src = q_nov + ((size_t)b * kN + n0 + row) * kD + half * 64;
    float v[64];
    float ss = 0.f;
#pragma unroll
    for (int i = 0; i < 16; ++i) {
      float4 f = reinterpret_cast<const float4*>(src)[i];
      v[i * 4 + 0] = f.x; v[i * 4 + 1] = f.y;
      v[i * 4 + 2] = f.z; v[i * 4 + 3] = f.w;
      ss += f.x * f.x + f.y * f.y + f.z * f.z + f.w * f.w;
    }
    ss += __shfl_xor(ss, 1, 32);
    const float scale = 1.0f / (sqrtf(ss) + kEps);
#pragma unroll
    for (int i = 0; i < 64; ++i)
      qn[row * 128 + half * 64 + i] = (bf16_t)(v[i] * scale);
  }
  __syncthreads();

  const int rt = wave * 16;
  v16bf a[4];
#pragma unroll
  for (int kk = 0; kk < 4; ++kk)
    a[kk] = load_frag_a(qn, 128, rt, kk * 32, lane);

  float rmax[8];
#pragma unroll
  for (int i = 0; i < 8; ++i) rmax[i] = -3.0e38f;

  const bf16_t* enb = en + (size_t)b * kM * kD;
  for (int mc = 0; mc < kM / 64; ++mc) {
#pragma unroll
    for (int ct = 0; ct < 4; ++ct) {
      v8f acc = {0.f, 0.f, 0.f, 0.f, 0.f, 0.f, 0.f, 0.f};
#pragma unroll
      for (int kk = 0; kk < 4; ++kk)
        acc = wmma_bf16(a[kk],
                        load_frag_b(enb, 128, mc * 64 + ct * 16, kk * 32, lane),
                        acc);
#pragma unroll
      for (int i = 0; i < 8; ++i) rmax[i] = fmaxf(rmax[i], acc[i]);
    }
  }
  // reduce max across the 16 lanes that hold different m-columns
#pragma unroll
  for (int i = 0; i < 8; ++i)
#pragma unroll
    for (int o = 1; o < 16; o <<= 1)
      rmax[i] = fmaxf(rmax[i], __shfl_xor(rmax[i], o, 16));
  if ((lane & 15) == 0) {
#pragma unroll
    for (int i = 0; i < 8; ++i) {
      const int n = n0 + rt + ((lane >> 4) << 3) + i;
      const float ws = w_nov[(size_t)b * kN + n] * surp[(size_t)b * kN + n];
      novelty[(size_t)b * kN + n] = ws * (1.0f - rmax[i]);
    }
  }
}

// ===========================================================================
// Kernel 4: top-32 per batch (descending, lower-index ties) + gather rows.
// ===========================================================================
__global__ __launch_bounds__(256) void k_topk(
    const float* __restrict__ novelty, const float* __restrict__ q_nov,
    const float* __restrict__ v_nov, float* __restrict__ candK,
    float* __restrict__ candV, float* __restrict__ candS) {
  __shared__ float nov[kN];
  __shared__ float rval[256];
  __shared__ int   ridx[256];
  __shared__ int   topidx[kCEM];
  const int b = blockIdx.x, tid = threadIdx.x;
  for (int i = tid; i < kN; i += 256) nov[i] = novelty[(size_t)b * kN + i];
  __syncthreads();
  for (int it = 0; it < kCEM; ++it) {
    float bv = -3.0e38f; int bi = 0x7fffffff;
    for (int i = tid; i < kN; i += 256) {
      const float v = nov[i];
      if (v > bv) { bv = v; bi = i; }
    }
    rval[tid] = bv; ridx[tid] = bi;
    __syncthreads();
    for (int s = 128; s > 0; s >>= 1) {
      if (tid < s) {
        const float ov = rval[tid + s]; const int oi = ridx[tid + s];
        if (ov > rval[tid] || (ov == rval[tid] && oi < ridx[tid])) {
          rval[tid] = ov; ridx[tid] = oi;
        }
      }
      __syncthreads();
    }
    if (tid == 0) {
      topidx[it] = ridx[0];
      candS[(size_t)b * kCEM + it] = rval[0];
      nov[ridx[0]] = -3.0e38f;
    }
    __syncthreads();
  }
  for (int e = tid; e < kCEM * kD; e += 256) {
    const int slot = e >> 7, d = e & 127;
    const int n = topidx[slot];
    candK[((size_t)b * kCEM + slot) * kD + d] = q_nov[((size_t)b * kN + n) * kD + d];
    candV[((size_t)b * kCEM + slot) * kD + d] = v_nov[((size_t)b * kN + n) * kD + d];
  }
}

// ===========================================================================
extern "C" void kernel_launch(void* const* d_in, const int* in_sizes, int n_in,
                              void* d_out, int out_size, void* d_ws,
                              size_t ws_size, hipStream_t stream) {
  const float* k_cand = (const float*)d_in[0];
  const float* v_cand = (const float*)d_in[1];
  const float* gate   = (const float*)d_in[2];
  const float* pm_K   = (const float*)d_in[3];
  const float* q_nov  = (const float*)d_in[4];
  const float* v_nov  = (const float*)d_in[5];
  const float* w_nov  = (const float*)d_in[6];
  const float* surp   = (const float*)d_in[7];
  const float* em_K   = (const float*)d_in[8];
  (void)in_sizes; (void)n_in; (void)out_size;

  float* out   = (float*)d_out;
  float* eligK = out;
  float* eligV = out + (size_t)kB * kR * kD;
  float* candK = out + (size_t)2 * kB * kR * kD;
  float* candV = candK + (size_t)kB * kCEM * kD;
  float* candS = candV + (size_t)kB * kCEM * kD;

  bf16_t* en      = (bf16_t*)d_ws;
  float*  novelty = (float*)((char*)d_ws + (size_t)kB * kM * kD * sizeof(bf16_t));
  (void)ws_size;

  // elig outputs are atomically accumulated -> zero them first (capturable).
  hipMemsetAsync(eligK, 0, (size_t)2 * kB * kR * kD * sizeof(float), stream);

  k_route_elig<<<kB * (kNC / 128), 256, 0, stream>>>(k_cand, v_cand, gate, pm_K,
                                                     eligK, eligV);
  k_em_norm<<<(kB * kM) / 8, 256, 0, stream>>>(em_K, en);
  k_novelty<<<kB * (kN / 128), 256, 0, stream>>>(q_nov, en, w_nov, surp, novelty);
  k_topk<<<kB, 256, 0, stream>>>(novelty, q_nov, v_nov, candK, candV, candS);
}
